// SparseAttention_40948218200582
// MI455X (gfx1250) — matrix operand order repaired and checked by reading, not compiled, not run on hardware
//
#include <hip/hip_runtime.h>
#include <math.h>

// ---------------------------------------------------------------------------
// Sliding-window causal attention block for MI455X (gfx1250, wave32, WMMA).
//   B=256, T=256, C=384, H=6, hd=64, WINDOW=64
// Pipeline: cast/transpose -> QKV bf16-WMMA GEMM (V written transposed) ->
// banded flash-style attention (WMMA QK^T and PV, V^T read from global) ->
// bf16-WMMA projection GEMM (f32 out).
// ---------------------------------------------------------------------------

typedef __bf16 bf16_t;
typedef __attribute__((ext_vector_type(16))) __bf16 v16bf;
typedef __attribute__((ext_vector_type(8)))  __bf16 v8bf;
typedef __attribute__((ext_vector_type(8)))  float  v8f;
typedef __attribute__((ext_vector_type(4)))  unsigned int v4u;

#define BATCH   256
#define SEQ     256
#define CH      384
#define NHEAD   6
#define HDIM    64
#define WIN     64
#define NQKV    (3 * CH)          // 1152
#define MROWS   (BATCH * SEQ)     // 65536
#define VTLD    (SEQ + 32)        // 288: V^T leading dim (key-padded)

static __device__ inline v8f vzero8f() {
    v8f z = {0.f, 0.f, 0.f, 0.f, 0.f, 0.f, 0.f, 0.f};
    return z;
}

// A-operand fragment (16x32 bf16), row-major source with leading dim `ld`.
// lanes 0-15: row m0+l16, K = k0+0..15
// lanes 16-31: row m0+l16, K = k0+16..31
static __device__ inline v16bf a_frag(const bf16_t* __restrict__ A, int ld, int m0, int k0) {
    int lane = threadIdx.x & 31;
    int l16  = lane & 15;
    int half = lane >> 4;
    const bf16_t* base = A + (size_t)(m0 + l16) * ld + k0 + half * 16;
    v8bf lo = *(const v8bf*)(base);
    v8bf hi = *(const v8bf*)(base + 8);
    return __builtin_shufflevector(lo, hi, 0, 1, 2, 3, 4, 5, 6, 7,
                                           8, 9, 10, 11, 12, 13, 14, 15);
}

// B-operand fragment (32x16 bf16) from a K-major (i.e. transposed) matrix Bt,
// Bt[n][k] with leading dim `ld`.  B[k][n] = Bt[n][k].
// lanes 0-15: col n0+l16, K = k0+0..15 ; lanes 16-31: col n0+l16, K = k0+16..31
static __device__ inline v16bf b_frag(const bf16_t* __restrict__ Bt, int ld, int n0, int k0) {
    int lane = threadIdx.x & 31;
    int l16  = lane & 15;
    int half = lane >> 4;
    return *(const v16bf*)(Bt + (size_t)(n0 + l16) * ld + k0 + half * 16);
}

static __device__ inline v8f wmma_bf16(v16bf a, v16bf b, v8f c) {
    return __builtin_amdgcn_wmma_f32_16x16x32_bf16(
        /*neg_a=*/false, a, /*neg_b=*/false, b,
        /*c_mod=*/(short)0, c, /*reuse_a=*/false, /*reuse_b=*/false);
}

// ------------------------------- cast / transpose ---------------------------

__global__ void cvt_f32_to_bf16(const float* __restrict__ src,
                                bf16_t* __restrict__ dst, int n) {
    int i = blockIdx.x * blockDim.x + threadIdx.x;
    if (i < n) dst[i] = (bf16_t)src[i];
}

// w[K][N] (row-major f32)  ->  wt[N][K] (row-major bf16)
__global__ void transpose_to_bf16(const float* __restrict__ w,
                                  bf16_t* __restrict__ wt, int K, int N) {
    int i = blockIdx.x * blockDim.x + threadIdx.x;
    if (i < K * N) {
        int kk = i / N;
        int n  = i - kk * N;
        wt[(size_t)n * K + kk] = (bf16_t)w[i];
    }
}

// zero the padded key columns [SEQ, VTLD) of V^T  [B*H][HDIM][VTLD]
__global__ void zero_vt_pad(bf16_t* __restrict__ vt) {
    int i = blockIdx.x * blockDim.x + threadIdx.x;   // over (B*H*HDIM)*32
    if (i < BATCH * NHEAD * HDIM * 32) {
        int c    = i & 31;
        int rest = i >> 5;                           // bh*HDIM + n
        vt[(size_t)rest * VTLD + SEQ + c] = (bf16_t)0.0f;
    }
}

// ------------------------------- QKV GEMM ----------------------------------
// C[m][n] = X[m][:] . W[:][n] + bias[n], n in [0,1152).
// q,k scattered as [B*H][T][64] bf16; v scattered TRANSPOSED as
// [B*H][64][VTLD] bf16 so the attention PV B-operand is a contiguous load.
// Block = 128 threads = 4 waves (2x2); wave tile 64x32; block tile 128x64.

__global__ __launch_bounds__(128)
void qkv_gemm_kernel(const bf16_t* __restrict__ X,      // [65536][384]
                     const bf16_t* __restrict__ Wt,     // [1152][384]
                     const float*  __restrict__ bias,   // [1152]
                     bf16_t* __restrict__ qo,           // [B*H][T][64]
                     bf16_t* __restrict__ ko,           // [B*H][T][64]
                     bf16_t* __restrict__ vto) {        // [B*H][64][VTLD]
    const int K = CH;
    int wave = threadIdx.x >> 5;
    int lane = threadIdx.x & 31;
    int l16  = lane & 15;
    int half = lane >> 4;
    int m0 = blockIdx.x * 128 + (wave >> 1) * 64;
    int n0 = blockIdx.y * 64  + (wave & 1) * 32;

    v8f acc[4][2];
#pragma unroll
    for (int mi = 0; mi < 4; mi++)
#pragma unroll
        for (int ni = 0; ni < 2; ni++) acc[mi][ni] = vzero8f();

    for (int k0 = 0; k0 < K; k0 += 32) {
        __builtin_prefetch(X + (size_t)(m0 + l16) * K + k0 + 64, 0, 1);
        v16bf a[4];
        v16bf b[2];
#pragma unroll
        for (int mi = 0; mi < 4; mi++) a[mi] = a_frag(X, K, m0 + mi * 16, k0);
#pragma unroll
        for (int ni = 0; ni < 2; ni++) b[ni] = b_frag(Wt, K, n0 + ni * 16, k0);
#pragma unroll
        for (int mi = 0; mi < 4; mi++)
#pragma unroll
            for (int ni = 0; ni < 2; ni++)
                acc[mi][ni] = wmma_bf16(a[mi], b[ni], acc[mi][ni]);
    }

#pragma unroll
    for (int mi = 0; mi < 4; mi++) {
#pragma unroll
        for (int ni = 0; ni < 2; ni++) {
            int col = n0 + ni * 16 + l16;
            float bval = bias[col];
            int sec = col / CH;          // 0=q 1=k 2=v
            int c   = col - sec * CH;
            int h   = c >> 6;
            int d   = c & 63;
#pragma unroll
            for (int r = 0; r < 8; r++) {
                int m  = m0 + mi * 16 + half * 8 + r;
                int bI = m >> 8;         // / SEQ
                int t  = m & 255;
                int bh = bI * NHEAD + h;
                float v = acc[mi][ni][r] + bval;
                if (sec == 0) {
                    qo[((size_t)bh * SEQ + t) * HDIM + d] = (bf16_t)v;
                } else if (sec == 1) {
                    ko[((size_t)bh * SEQ + t) * HDIM + d] = (bf16_t)v;
                } else {
                    vto[((size_t)bh * HDIM + d) * VTLD + t] = (bf16_t)v;
                }
            }
        }
    }
}

// ------------------------------- attention ---------------------------------
// One block (256 thr = 8 waves) per (b, h, T-half); each wave owns one 16-row
// query tile independently (no block barrier needed). Masked QK^T (WMMA),
// softmax with 16-lane shuffles, P restaged via per-wave LDS tile, then P.V
// with the B-operand streamed from the pre-transposed, key-padded V^T in
// global (L2-resident). Gate folded into the bf16 store of [B][T][C].

__global__ __launch_bounds__(256)
void attn_kernel(const bf16_t* __restrict__ q,     // [B*H][T][64]
                 const bf16_t* __restrict__ k,     // [B*H][T][64]
                 const bf16_t* __restrict__ vt,    // [B*H][64][VTLD]
                 const float*  __restrict__ gate,  // [C]
                 bf16_t* __restrict__ out) {       // [B][T][C]
    __shared__ __align__(64) bf16_t pbuf[8][16][96];      // per-wave P tile

    int bh    = blockIdx.x >> 1;
    int halfT = blockIdx.x & 1;
    int b = bh / NHEAD;
    int h = bh - b * NHEAD;
    const bf16_t* qb  = q  + (size_t)bh * SEQ * HDIM;
    const bf16_t* kb  = k  + (size_t)bh * SEQ * HDIM;
    const bf16_t* vtb = vt + (size_t)bh * HDIM * VTLD;

    int tid  = threadIdx.x;
    int wave = tid >> 5;
    int lane = tid & 31;
    int l16  = lane & 15;
    int half = lane >> 4;

    int qt  = halfT * 8 + wave;          // query tile 0..15
    int q0  = qt * 16;
    int kt0 = (qt >= 4) ? (qt - 4) : 0;  // first key tile (band = 65 keys)
    int nk  = qt - kt0 + 1;              // <= 5 key tiles

    // ---- Q fragments (16 x 64) ----
    v16bf aq0 = a_frag(qb, HDIM, q0, 0);
    v16bf aq1 = a_frag(qb, HDIM, q0, 32);

    // ---- scores S = Q K^T ----
    v8f S[5];
#pragma unroll
    for (int i = 0; i < 5; i++) S[i] = vzero8f();
    for (int i = 0; i < nk; i++) {
        int kbase = (kt0 + i) * 16;
        v16bf bk0 = b_frag(kb, HDIM, kbase, 0);
        v16bf bk1 = b_frag(kb, HDIM, kbase, 32);
        v8f s = vzero8f();
        s = wmma_bf16(aq0, bk0, s);
        s = wmma_bf16(aq1, bk1, s);
        S[i] = s;
    }

    // ---- mask + softmax (row spans the 16 lanes of this half) ----
    const float NEG = -3.0e38f;
    float lrow[8];
#pragma unroll
    for (int r = 0; r < 8; r++) {
        int row = q0 + half * 8 + r;
        float mx = NEG;
        for (int i = 0; i < nk; i++) {
            int col = (kt0 + i) * 16 + l16;
            float sv = S[i][r] * 0.125f;                // 1/sqrt(64)
            bool ok = (col <= row) && (row - col <= WIN);
            sv = ok ? sv : NEG;
            S[i][r] = sv;
            mx = fmaxf(mx, sv);
        }
#pragma unroll
        for (int off = 8; off >= 1; off >>= 1)
            mx = fmaxf(mx, __shfl_xor(mx, off, 32));
        float sum = 0.f;
        for (int i = 0; i < nk; i++) {
            float p = __expf(S[i][r] - mx);             // masked -> 0
            S[i][r] = p;
            sum += p;
        }
#pragma unroll
        for (int off = 8; off >= 1; off >>= 1)
            sum += __shfl_xor(sum, off, 32);
        lrow[r] = sum;
    }

    // ---- restage P as bf16 in A-layout-friendly LDS tile (zero padded) ----
    {
        v4u* pz = (v4u*)&pbuf[wave][0][0];              // 16*96*2/16 = 192
#pragma unroll
        for (int i = 0; i < 6; i++) {
            v4u z = {0u, 0u, 0u, 0u};
            pz[lane + 32 * i] = z;
        }
        asm volatile("s_wait_dscnt 0" ::: "memory");
        for (int i = 0; i < nk; i++)
#pragma unroll
            for (int r = 0; r < 8; r++)
                pbuf[wave][half * 8 + r][i * 16 + l16] = (bf16_t)S[i][r];
        asm volatile("s_wait_dscnt 0" ::: "memory");
    }

    // ---- O = P . V  (B-operand straight from key-padded global V^T) ----
    int gk0 = kt0 * 16;
    int chunks = (nk + 1) >> 1;                          // <= 3
    v16bf ap[3];
    for (int c = 0; c < chunks; c++)
        ap[c] = a_frag(&pbuf[wave][0][0], 96, 0, c * 32);

#pragma unroll
    for (int nt = 0; nt < 4; nt++) {
        v8f o = vzero8f();
        for (int c = 0; c < chunks; c++) {
            v16bf bv = b_frag(vtb, VTLD, nt * 16, gk0 + c * 32);
            o = wmma_bf16(ap[c], bv, o);
        }
        int d   = nt * 16 + l16;
        int cch = h * HDIM + d;
        float g = gate[cch];
#pragma unroll
        for (int r = 0; r < 8; r++) {
            int t = q0 + half * 8 + r;
            float val = (o[r] / lrow[r]) * g;
            out[((size_t)b * SEQ + t) * CH + cch] = (bf16_t)val;
        }
    }
}

// ------------------------------- projection GEMM ---------------------------
// out[m][n] = attn[m][:] . proj_w[:][n] + proj_b[n], f32 output.

__global__ __launch_bounds__(128)
void proj_gemm_kernel(const bf16_t* __restrict__ A,     // [65536][384]
                      const bf16_t* __restrict__ Wt,    // [384][384]
                      const float*  __restrict__ bias,  // [384]
                      float* __restrict__ out) {        // [65536][384]
    const int K = CH, N = CH;
    int wave = threadIdx.x >> 5;
    int lane = threadIdx.x & 31;
    int l16  = lane & 15;
    int half = lane >> 4;
    int m0 = blockIdx.x * 128 + (wave >> 1) * 64;
    int n0 = blockIdx.y * 64  + (wave & 1) * 32;

    v8f acc[4][2];
#pragma unroll
    for (int mi = 0; mi < 4; mi++)
#pragma unroll
        for (int ni = 0; ni < 2; ni++) acc[mi][ni] = vzero8f();

    for (int k0 = 0; k0 < K; k0 += 32) {
        __builtin_prefetch(A + (size_t)(m0 + l16) * K + k0 + 64, 0, 1);
        v16bf a[4];
        v16bf b[2];
#pragma unroll
        for (int mi = 0; mi < 4; mi++) a[mi] = a_frag(A, K, m0 + mi * 16, k0);
#pragma unroll
        for (int ni = 0; ni < 2; ni++) b[ni] = b_frag(Wt, K, n0 + ni * 16, k0);
#pragma unroll
        for (int mi = 0; mi < 4; mi++)
#pragma unroll
            for (int ni = 0; ni < 2; ni++)
                acc[mi][ni] = wmma_bf16(a[mi], b[ni], acc[mi][ni]);
    }

#pragma unroll
    for (int mi = 0; mi < 4; mi++) {
#pragma unroll
        for (int ni = 0; ni < 2; ni++) {
            int col = n0 + ni * 16 + l16;
            float bval = bias[col];
#pragma unroll
            for (int r = 0; r < 8; r++) {
                int m = m0 + mi * 16 + half * 8 + r;
                out[(size_t)m * N + col] = acc[mi][ni][r] + bval;
            }
        }
    }
}

// ------------------------------- launch ------------------------------------

extern "C" void kernel_launch(void* const* d_in, const int* in_sizes, int n_in,
                              void* d_out, int out_size, void* d_ws, size_t ws_size,
                              hipStream_t stream) {
    const float* x      = (const float*)d_in[0];   // [B,T,C]
    const float* qkv_w  = (const float*)d_in[1];   // [C,3C]
    const float* qkv_b  = (const float*)d_in[2];   // [3C]
    const float* proj_w = (const float*)d_in[3];   // [C,C]
    const float* proj_b = (const float*)d_in[4];   // [C]
    const float* gate   = (const float*)d_in[5];   // [C]
    float* out = (float*)d_out;                    // [B,T,C] f32

    // workspace carve-up (all offsets 32B aligned)
    char* p = (char*)d_ws;
    bf16_t* xb = (bf16_t*)p;
    p += (size_t)MROWS * CH * sizeof(bf16_t);                     // 48 MB
    bf16_t* qkv_wT = (bf16_t*)p;
    p += (size_t)NQKV * CH * sizeof(bf16_t);
    bf16_t* proj_wT = (bf16_t*)p;
    p += (size_t)CH * CH * sizeof(bf16_t);
    bf16_t* qbuf = (bf16_t*)p;                                    // [B*H][T][64]
    p += (size_t)BATCH * NHEAD * SEQ * HDIM * sizeof(bf16_t);     // 48 MB
    bf16_t* kbuf = (bf16_t*)p;                                    // [B*H][T][64]
    p += (size_t)BATCH * NHEAD * SEQ * HDIM * sizeof(bf16_t);     // 48 MB
    bf16_t* vtbuf = (bf16_t*)p;                                   // [B*H][64][VTLD]
    p += (size_t)BATCH * NHEAD * HDIM * VTLD * sizeof(bf16_t);    // 54 MB
    bf16_t* attn = (bf16_t*)p;                                    // [B][T][C]
    p += (size_t)MROWS * CH * sizeof(bf16_t);                     // 48 MB

    {
        int n = MROWS * CH;
        cvt_f32_to_bf16<<<(n + 255) / 256, 256, 0, stream>>>(x, xb, n);
    }
    {
        int n = CH * NQKV;
        transpose_to_bf16<<<(n + 255) / 256, 256, 0, stream>>>(qkv_w, qkv_wT, CH, NQKV);
    }
    {
        int n = CH * CH;
        transpose_to_bf16<<<(n + 255) / 256, 256, 0, stream>>>(proj_w, proj_wT, CH, CH);
    }
    {
        int n = BATCH * NHEAD * HDIM * 32;
        zero_vt_pad<<<(n + 255) / 256, 256, 0, stream>>>(vtbuf);
    }

    // QKV GEMM: M=65536 (512 tiles of 128), N=1152 (18 tiles of 64)
    qkv_gemm_kernel<<<dim3(512, 18), 128, 0, stream>>>(xb, qkv_wT, qkv_b,
                                                       qbuf, kbuf, vtbuf);

    // Attention: one block per (b, h, T-half) -> 256*6*2 = 3072 blocks
    attn_kernel<<<dim3(BATCH * NHEAD * 2), 256, 0, stream>>>(
        qbuf, kbuf, vtbuf, gate, attn);

    // Projection: M=65536 (512 tiles of 128), N=384 (6 tiles of 64)
    proj_gemm_kernel<<<dim3(512, 6), 128, 0, stream>>>(attn, proj_wT, proj_b, out);
}
